// InfiniSpanAttention_76716705841652
// MI455X (gfx1250) — matrix operand order repaired and checked by reading, not compiled, not run on hardware
//
#include <hip/hip_runtime.h>

#define DIMSZ   1024
#define NHEADS  16
#define HDIM    64
#define LLOC    2048
#define LGLB    512
#define NBATCH  2
#define QKSCALE 0.125f   // 64^-0.5

typedef __attribute__((ext_vector_type(16))) __bf16 bf16x16;
typedef __attribute__((ext_vector_type(8)))  float  f32x8;
typedef __attribute__((ext_vector_type(4)))  unsigned int u32x4;
typedef __attribute__((ext_vector_type(8)))  int i32x8;
typedef __attribute__((ext_vector_type(4)))  int i32x4;

struct alignas(16) U4 { unsigned int x, y, z, w; };

union Frag {
  bf16x16 v;
  U4 q[2];
  __device__ Frag() {}
};

// wave-local LDS ordering fence (store->load through same wave's LDS buffer)
#define DS_FENCE() asm volatile("s_wait_dscnt 0" ::: "memory")

__device__ __forceinline__ unsigned short f2bf(float f) {
  unsigned int u = __float_as_uint(f);
  u += 0x7FFFu + ((u >> 16) & 1u);          // round-to-nearest-even
  return (unsigned short)(u >> 16);
}

// 16-lane (half-wave) reductions via ds_swizzle xor patterns (wave32-safe).
__device__ __forceinline__ float red_max16(float v) {
  v = fmaxf(v, __int_as_float(__builtin_amdgcn_ds_swizzle(__float_as_int(v), 0x041f)));
  v = fmaxf(v, __int_as_float(__builtin_amdgcn_ds_swizzle(__float_as_int(v), 0x081f)));
  v = fmaxf(v, __int_as_float(__builtin_amdgcn_ds_swizzle(__float_as_int(v), 0x101f)));
  v = fmaxf(v, __int_as_float(__builtin_amdgcn_ds_swizzle(__float_as_int(v), 0x201f)));
  return v;
}
__device__ __forceinline__ float red_sum16(float v) {
  v += __int_as_float(__builtin_amdgcn_ds_swizzle(__float_as_int(v), 0x041f));
  v += __int_as_float(__builtin_amdgcn_ds_swizzle(__float_as_int(v), 0x081f));
  v += __int_as_float(__builtin_amdgcn_ds_swizzle(__float_as_int(v), 0x101f));
  v += __int_as_float(__builtin_amdgcn_ds_swizzle(__float_as_int(v), 0x201f));
  return v;
}

__device__ __forceinline__ f32x8 wmma_bf16(const Frag& a, const Frag& b, f32x8 c) {
  return __builtin_amdgcn_wmma_f32_16x16x32_bf16(false, a.v, false, b.v,
                                                 (short)0, c, false, false);
}

#if defined(__gfx1250__) && __has_builtin(__builtin_amdgcn_tensor_load_to_lds) && \
    __has_builtin(__builtin_amdgcn_s_wait_tensorcnt)
#define USE_TDM 1
// Issue a TDM DMA of a 64-row x 32-col bf16 tile of W (row stride 1024 elems)
// into LDS at byte offset lds_addr. D# built per CDNA5 ISA 8.3/8.4.
// This toolchain declares the 6-arg builtin:
//   (u32x4 group0, i32x8 group1, i32x4 group2, i32x4 group3, i32x8 extra, i32 cpol)
__device__ __forceinline__ void tdm_load_tile(unsigned lds_addr,
                                              const unsigned short* gsrc) {
  unsigned long long ga = (unsigned long long)gsrc;
  u32x4 g0 = { 1u,                                   // count=1 valid descriptor
               lds_addr,                              // LDS byte address
               (unsigned)ga,                          // global addr [31:0]
               (unsigned)(ga >> 32) | 0x80000000u };  // addr[56:32] | type=2
  i32x8 g1 = { 0x10000,                    // data_size=1 (2 bytes/elem)
               0x04000000,                 // tensor_dim0 = 1024 (bits 63:48)
               0x04000000,                 // tensor_dim1 = 1024 (bits 95:80)
               0x00200000,                 // tile_dim0 = 32 (bits 127:112)
               64,                         // tile_dim1 = 64 (bits 143:128)
               1024,                       // tensor_dim0_stride = 1024
               0, 0 };
  i32x4 z4 = { 0, 0, 0, 0 };
  i32x8 z8 = { 0, 0, 0, 0, 0, 0, 0, 0 };
  __builtin_amdgcn_tensor_load_to_lds(g0, g1, z4, z4, z8, 0);
}
#else
#define USE_TDM 0
#endif

// ---------------- fp32 -> bf16 convert ----------------
__global__ __launch_bounds__(256) void cvt_bf16_kernel(const float* __restrict__ src,
                                                       unsigned short* __restrict__ dst,
                                                       int n) {
  int i = (blockIdx.x * 256 + threadIdx.x) * 4;
  if (i < n) {
    float x0 = src[i], x1 = src[i + 1], x2 = src[i + 2], x3 = src[i + 3];
    unsigned long long p = (unsigned long long)f2bf(x0)
                         | ((unsigned long long)f2bf(x1) << 16)
                         | ((unsigned long long)f2bf(x2) << 32)
                         | ((unsigned long long)f2bf(x3) << 48);
    *(unsigned long long*)(dst + i) = p;
  }
}

// ---------------- Y = X @ W^T + bias (bf16 WMMA, f32 accum) ----------------
// mode 0: Ybf bf16 [B,H,Sseq,64]      (Q, K layouts)
// mode 1: Ybf bf16 [B,H,64,Sseq]      (transposed V layout)
// mode 2: Yf  f32  [S_total,1024]     (final output)
__global__ __launch_bounds__(256) void gemm_bf16_kernel(
    const unsigned short* __restrict__ X,     // [S_total, 1024] bf16
    const unsigned short* __restrict__ W,     // [1024, 1024] bf16 row-major (out, in)
    const float* __restrict__ bias,           // [1024]
    unsigned short* __restrict__ Ybf,
    float* __restrict__ Yf,
    int Sseq, int mode) {
  __shared__ alignas(16) unsigned short ldsB[2][64 * 32];   // double-buffered W tile

  const int n0   = blockIdx.x * 64;
  const int m0   = blockIdx.y * 128;
  const int tid  = threadIdx.x;
  const int wave = tid >> 5, lane = tid & 31;
  const int hi   = lane >> 4, l15 = lane & 15;

  f32x8 acc[4];
#pragma unroll
  for (int j = 0; j < 4; ++j)
#pragma unroll
    for (int r = 0; r < 8; ++r) acc[j][r] = 0.f;

  const long xrow = (long)(m0 + wave * 16 + l15) * DIMSZ;

#if USE_TDM
  const unsigned short* wtile = W + (long)n0 * DIMSZ;   // tile row origin, col 0
  unsigned ldsaddr[2];
  ldsaddr[0] = (unsigned)(unsigned long long)(void*)&ldsB[0][0];
  ldsaddr[1] = (unsigned)(unsigned long long)(void*)&ldsB[1][0];
  if (wave == 0) tdm_load_tile(ldsaddr[0], wtile);       // tile for kt=0

#pragma clang loop unroll(disable)
  for (int kt = 0; kt < DIMSZ / 32; ++kt) {
    const int cur = kt & 1;
    if (wave == 0) __builtin_amdgcn_s_wait_tensorcnt((short)0);  // tile kt done
    __syncthreads();
    if (wave == 0 && kt + 1 < DIMSZ / 32)                // DMA next tile while
      tdm_load_tile(ldsaddr[cur ^ 1], wtile + (kt + 1) * 32);  // computing this one
    const int k0 = kt * 32;

    Frag a;
    a.q[0] = *(const U4*)(X + xrow + k0 + hi * 8);
    a.q[1] = *(const U4*)(X + xrow + k0 + hi * 8 + 16);

#pragma unroll
    for (int j = 0; j < 4; ++j) {
      Frag b;
      const unsigned short* p = ldsB[cur] + (j * 16 + l15) * 32 + hi * 8;
      b.q[0] = *(const U4*)p;
      b.q[1] = *(const U4*)(p + 16);
      acc[j] = wmma_bf16(a, b, acc[j]);
    }
  }
#else
  const int wr = tid >> 2;          // 0..63 (W row within N tile)
  const int wc = (tid & 3) * 8;     // 0,8,16,24
  {
    U4 st = *(const U4*)(W + (long)(n0 + wr) * DIMSZ + wc);
    *(U4*)(ldsB[0] + wr * 32 + wc) = st;
  }
  __syncthreads();

#pragma clang loop unroll(disable)
  for (int kt = 0; kt < DIMSZ / 32; ++kt) {
    const int cur = kt & 1;
    const int k0 = kt * 32;
    const bool have = (kt + 1) < DIMSZ / 32;
    U4 nxt;
    if (have) {
      nxt = *(const U4*)(W + (long)(n0 + wr) * DIMSZ + k0 + 32 + wc);
      __builtin_prefetch(W + (long)(n0 + wr) * DIMSZ + k0 + 64 + wc, 0, 1);
    }

    Frag a;
    a.q[0] = *(const U4*)(X + xrow + k0 + hi * 8);
    a.q[1] = *(const U4*)(X + xrow + k0 + hi * 8 + 16);

#pragma unroll
    for (int j = 0; j < 4; ++j) {
      Frag b;
      const unsigned short* p = ldsB[cur] + (j * 16 + l15) * 32 + hi * 8;
      b.q[0] = *(const U4*)p;
      b.q[1] = *(const U4*)(p + 16);
      acc[j] = wmma_bf16(a, b, acc[j]);
    }
    if (have) *(U4*)(ldsB[cur ^ 1] + wr * 32 + wc) = nxt;
    __syncthreads();
  }
#endif

  const int sbase = m0 + wave * 16 + hi * 8;   // 8 consecutive rows, same batch
  const int bb    = sbase / Sseq;
  const int ss    = sbase - bb * Sseq;

#pragma unroll
  for (int j = 0; j < 4; ++j) {
    const int   o  = n0 + j * 16 + l15;
    const float bv = bias[o];
    const int   h  = o >> 6, d = o & 63;
    if (mode == 0) {
      unsigned short* dst = Ybf + ((long)(bb * NHEADS + h) * Sseq + ss) * HDIM + d;
#pragma unroll
      for (int r = 0; r < 8; ++r) dst[(long)r * HDIM] = f2bf(acc[j][r] + bv);
    } else if (mode == 1) {
      U4 pk;
      pk.x = (unsigned int)f2bf(acc[j][0] + bv) | ((unsigned int)f2bf(acc[j][1] + bv) << 16);
      pk.y = (unsigned int)f2bf(acc[j][2] + bv) | ((unsigned int)f2bf(acc[j][3] + bv) << 16);
      pk.z = (unsigned int)f2bf(acc[j][4] + bv) | ((unsigned int)f2bf(acc[j][5] + bv) << 16);
      pk.w = (unsigned int)f2bf(acc[j][6] + bv) | ((unsigned int)f2bf(acc[j][7] + bv) << 16);
      *(U4*)(Ybf + ((long)(bb * NHEADS + h) * HDIM + d) * Sseq + ss) = pk;
    } else {
      float* dst = Yf + (long)sbase * DIMSZ + o;
#pragma unroll
      for (int r = 0; r < 8; ++r) dst[(long)r * DIMSZ] = acc[j][r] + bv;
    }
  }
}

// ---------------- fused dual-span flash attention ----------------
__global__ __launch_bounds__(128) void attn_kernel(
    const unsigned short* __restrict__ Q,    // [B,H,L,64]
    const unsigned short* __restrict__ Kl,   // [B,H,L,64]
    const unsigned short* __restrict__ Vlt,  // [B,H,64,L]
    const unsigned short* __restrict__ Kg,   // [B,H,G,64]
    const unsigned short* __restrict__ Vgt,  // [B,H,64,G]
    const float* __restrict__ gatep,
    unsigned short* __restrict__ Out) {      // [B,L,1024] bf16
  __shared__ alignas(16) unsigned short ldsP[4][16 * 32];   // per-wave P buffer

  const int qt = blockIdx.x, h = blockIdx.y, b = blockIdx.z;
  const int tid  = threadIdx.x;
  const int wave = tid >> 5, lane = tid & 31;
  const int hi   = lane >> 4, l15 = lane & 15;
  const int q0   = qt * 64 + wave * 16;
  const long bh  = (long)(b * NHEADS + h);
  unsigned short* myP = ldsP[wave];

  // Q fragments for the two K=32 steps over hd=64
  Frag qa0, qa1;
  {
    const unsigned short* p = Q + (bh * LLOC + q0 + l15) * HDIM + hi * 8;
    qa0.q[0] = *(const U4*)(p);
    qa0.q[1] = *(const U4*)(p + 16);
    qa1.q[0] = *(const U4*)(p + 32);
    qa1.q[1] = *(const U4*)(p + 48);
  }
  const float gv = *gatep;
  const float g  = 1.f / (1.f + __expf(-gv));

  f32x8 res[4];
#pragma unroll
  for (int j = 0; j < 4; ++j)
#pragma unroll
    for (int r = 0; r < 8; ++r) res[j][r] = 0.f;

  for (int phase = 0; phase < 2; ++phase) {
    const unsigned short* K  = phase ? Kg  : Kl;
    const unsigned short* Vt = phase ? Vgt : Vlt;
    const int   Lk  = phase ? LGLB : LLOC;
    const float wgt = phase ? (1.f - g) : g;

    f32x8 acc[4];
    float mrow[8], lrow[8];
#pragma unroll
    for (int j = 0; j < 4; ++j)
#pragma unroll
      for (int r = 0; r < 8; ++r) acc[j][r] = 0.f;
#pragma unroll
    for (int r = 0; r < 8; ++r) { mrow[r] = -3.0e38f; lrow[r] = 0.f; }

#pragma clang loop unroll(disable)
    for (int kc = 0; kc < Lk; kc += 32) {
      // scores for 32 keys: two 16x16 C tiles, each accumulated over hd=64
      f32x8 s0, s1;
      {
        Frag kb0, kb1;
        const unsigned short* p = K + (bh * Lk + kc + l15) * HDIM + hi * 8;
        kb0.q[0] = *(const U4*)p;        kb0.q[1] = *(const U4*)(p + 16);
        kb1.q[0] = *(const U4*)(p + 32); kb1.q[1] = *(const U4*)(p + 48);
        f32x8 z;
#pragma unroll
        for (int r = 0; r < 8; ++r) z[r] = 0.f;
        z  = wmma_bf16(qa0, kb0, z);
        s0 = wmma_bf16(qa1, kb1, z);
      }
      {
        Frag kb0, kb1;
        const unsigned short* p = K + (bh * Lk + kc + 16 + l15) * HDIM + hi * 8;
        kb0.q[0] = *(const U4*)p;        kb0.q[1] = *(const U4*)(p + 16);
        kb1.q[0] = *(const U4*)(p + 32); kb1.q[1] = *(const U4*)(p + 48);
        f32x8 z;
#pragma unroll
        for (int r = 0; r < 8; ++r) z[r] = 0.f;
        z  = wmma_bf16(qa0, kb0, z);
        s1 = wmma_bf16(qa1, kb1, z);
      }

      // online softmax update; write P tile to LDS (C layout -> row-major)
      float alpha[8];
#pragma unroll
      for (int r = 0; r < 8; ++r) {
        float a0 = s0[r] * QKSCALE, a1 = s1[r] * QKSCALE;
        float mx = red_max16(fmaxf(a0, a1));
        float mn = fmaxf(mrow[r], mx);
        float al = __expf(mrow[r] - mn);
        float p0 = __expf(a0 - mn), p1 = __expf(a1 - mn);
        lrow[r] = lrow[r] * al + red_sum16(p0 + p1);
        mrow[r] = mn;
        alpha[r] = al;
        const int row = hi * 8 + r;
        myP[row * 32 + l15]      = f2bf(p0);
        myP[row * 32 + 16 + l15] = f2bf(p1);
      }
#pragma unroll
      for (int j = 0; j < 4; ++j)
#pragma unroll
        for (int r = 0; r < 8; ++r) acc[j][r] *= alpha[r];

      // wave-private LDS buffer: same-wave DS ops execute in order, so a
      // DS-count fence (not a workgroup barrier) is enough for the transpose.
      DS_FENCE();
      Frag pa;
      pa.q[0] = *(const U4*)(myP + l15 * 32 + hi * 8);
      pa.q[1] = *(const U4*)(myP + l15 * 32 + hi * 8 + 16);
      DS_FENCE();

      // out += P (16x32) @ V (32x64); V^T layout makes B fragments contiguous
#pragma unroll
      for (int j = 0; j < 4; ++j) {
        Frag vb;
        const unsigned short* p = Vt + (bh * HDIM + j * 16 + l15) * Lk + kc + hi * 8;
        vb.q[0] = *(const U4*)p;
        vb.q[1] = *(const U4*)(p + 16);
        acc[j] = wmma_bf16(pa, vb, acc[j]);
      }
    }

#pragma unroll
    for (int r = 0; r < 8; ++r) {
      const float inv = wgt / lrow[r];
#pragma unroll
      for (int j = 0; j < 4; ++j) res[j][r] += acc[j][r] * inv;
    }
  }

  // store gated result in [B, L, 1024] bf16 for the output projection
#pragma unroll
  for (int j = 0; j < 4; ++j) {
    const int o = h * HDIM + j * 16 + l15;
    unsigned short* dst = Out + ((long)b * LLOC + q0 + hi * 8) * DIMSZ + o;
#pragma unroll
    for (int r = 0; r < 8; ++r) dst[(long)r * DIMSZ] = f2bf(res[j][r]);
  }
}

extern "C" void kernel_launch(void* const* d_in, const int* in_sizes, int n_in,
                              void* d_out, int out_size, void* d_ws, size_t ws_size,
                              hipStream_t stream) {
  const float* lh_f = (const float*)d_in[0];
  const float* gh_f = (const float*)d_in[1];
  const float* qw  = (const float*)d_in[2];
  const float* qbv = (const float*)d_in[3];
  const float* kw  = (const float*)d_in[4];
  const float* kbv = (const float*)d_in[5];
  const float* vw  = (const float*)d_in[6];
  const float* vbv = (const float*)d_in[7];
  const float* kgw = (const float*)d_in[8];
  const float* kgb = (const float*)d_in[9];
  const float* vgw = (const float*)d_in[10];
  const float* vgb = (const float*)d_in[11];
  const float* ow  = (const float*)d_in[12];
  const float* obv = (const float*)d_in[13];
  const float* gate = (const float*)d_in[14];

  unsigned short* ws = (unsigned short*)d_ws;
  size_t off = 0;
  auto take = [&](size_t n) { unsigned short* p = ws + off; off += n; return p; };
  const size_t nLH = (size_t)NBATCH * LLOC * DIMSZ;   // 4,194,304
  const size_t nGH = (size_t)NBATCH * LGLB * DIMSZ;   // 1,048,576
  const size_t nW  = (size_t)DIMSZ * DIMSZ;           // 1,048,576

  unsigned short* lh   = take(nLH);
  unsigned short* gh   = take(nGH);
  unsigned short* wq   = take(nW);
  unsigned short* wk   = take(nW);
  unsigned short* wv   = take(nW);
  unsigned short* wkg  = take(nW);
  unsigned short* wvg  = take(nW);
  unsigned short* wo   = take(nW);
  unsigned short* Qb   = take(nLH);
  unsigned short* Kb   = take(nLH);
  unsigned short* Vtb  = take(nLH);
  unsigned short* Kgb  = take(nGH);
  unsigned short* Vgtb = take(nGH);
  unsigned short* AO   = take(nLH);

  cvt_bf16_kernel<<<(unsigned)(nLH / 1024), 256, 0, stream>>>(lh_f, lh, (int)nLH);
  cvt_bf16_kernel<<<(unsigned)(nGH / 1024), 256, 0, stream>>>(gh_f, gh, (int)nGH);
  cvt_bf16_kernel<<<(unsigned)(nW / 1024), 256, 0, stream>>>(qw,  wq,  (int)nW);
  cvt_bf16_kernel<<<(unsigned)(nW / 1024), 256, 0, stream>>>(kw,  wk,  (int)nW);
  cvt_bf16_kernel<<<(unsigned)(nW / 1024), 256, 0, stream>>>(vw,  wv,  (int)nW);
  cvt_bf16_kernel<<<(unsigned)(nW / 1024), 256, 0, stream>>>(kgw, wkg, (int)nW);
  cvt_bf16_kernel<<<(unsigned)(nW / 1024), 256, 0, stream>>>(vgw, wvg, (int)nW);
  cvt_bf16_kernel<<<(unsigned)(nW / 1024), 256, 0, stream>>>(ow,  wo,  (int)nW);

  dim3 gL(DIMSZ / 64, (NBATCH * LLOC) / 128);   // (16, 32)
  dim3 gG(DIMSZ / 64, (NBATCH * LGLB) / 128);   // (16, 8)
  gemm_bf16_kernel<<<gL, 256, 0, stream>>>(lh, wq,  qbv, Qb,   nullptr, LLOC, 0);
  gemm_bf16_kernel<<<gL, 256, 0, stream>>>(lh, wk,  kbv, Kb,   nullptr, LLOC, 0);
  gemm_bf16_kernel<<<gL, 256, 0, stream>>>(lh, wv,  vbv, Vtb,  nullptr, LLOC, 1);
  gemm_bf16_kernel<<<gG, 256, 0, stream>>>(gh, wkg, kgb, Kgb,  nullptr, LGLB, 0);
  gemm_bf16_kernel<<<gG, 256, 0, stream>>>(gh, wvg, vgb, Vgtb, nullptr, LGLB, 1);

  attn_kernel<<<dim3(LLOC / 64, NHEADS, NBATCH), 128, 0, stream>>>(
      Qb, Kb, Vtb, Kgb, Vgtb, gate, AO);

  gemm_bf16_kernel<<<gL, 256, 0, stream>>>(AO, wo, obv, nullptr, (float*)d_out, LLOC, 2);
}
